// LinformerSelfAttention_85280870629829
// MI455X (gfx1250) — matrix operand order repaired and checked by reading, not compiled, not run on hardware
//
#include <hip/hip_runtime.h>

// ---------------- problem constants ----------------
constexpr int CB  = 4;      // batch
constexpr int CS  = 4096;   // seq
constexpr int CD  = 1024;   // model dim
constexpr int CH  = 16;     // heads
constexpr int CKP = 256;    // linformer K
constexpr int CDH = 64;     // head dim
constexpr int NTOK = CB * CS;          // 16384

typedef __bf16 bf16;
typedef __attribute__((ext_vector_type(16))) __bf16 v16bf;
typedef __attribute__((ext_vector_type(8)))  __bf16 v8bf;
typedef __attribute__((ext_vector_type(8)))  float  v8f;

union Frag16 { v16bf v; v8bf h[2]; };

// load a 16-element bf16 fragment from two 16B-aligned 8-element halves
__device__ __forceinline__ v16bf load_frag(const bf16* p0, const bf16* p1) {
  Frag16 f;
  f.h[0] = *(const v8bf*)p0;
  f.h[1] = *(const v8bf*)p1;
  return f.v;
}

__device__ __forceinline__ v8f wmma_bf16(v16bf a, v16bf b, v8f c) {
  return __builtin_amdgcn_wmma_f32_16x16x32_bf16(
      /*neg_a=*/false, a, /*neg_b=*/false, b,
      /*c_mod=*/(short)0, c, /*reuse_a=*/false, /*reuse_b=*/false);
}

// ---- CDNA5 async global->LDS copy (16B per lane), tracked by ASYNCcnt ----
__device__ __forceinline__ void async_copy_b128(unsigned lds_byte_addr,
                                                const void* gaddr) {
  asm volatile("global_load_async_to_lds_b128 %0, %1, off"
               :: "v"(lds_byte_addr), "v"(gaddr)
               : "memory");
}
#define ASYNC_WAIT(n) asm volatile("s_wait_asynccnt " #n ::: "memory")

// ---------------- stage 0: casts ----------------
__global__ void cast_bf16_kernel(const float* __restrict__ in,
                                 bf16* __restrict__ out, int n) {
  int i = blockIdx.x * blockDim.x + threadIdx.x;
  int stride = gridDim.x * blockDim.x;
  for (; i < n; i += stride) out[i] = (bf16)in[i];
}

// k-blocked transposed weight: wb[(k/32)*D*32 + n*32 + k%32] = W[n][k]
// (so a 64(n) x 32(k) B-tile is one contiguous 4KB chunk in [n][k] order)
__global__ void transpose_block_kernel(const float* __restrict__ in,
                                       bf16* __restrict__ out) {
  int o = blockIdx.x * blockDim.x + threadIdx.x;   // o = k*1024 + n
  int k = o >> 10;
  int n = o & 1023;
  out[((size_t)(k >> 5) << 15) + (n << 5) + (k & 31)] = (bf16)in[n * CD + k];
}

// ---------------- stage 1: QKV projection ----------------
// C[tok, c] = sum_i hsb[tok, i] * W[c, i] ; +bias, mask.
// z==0 -> q stored row-major [B,H,S,DH]; z==1/2 -> k/v stored k-blocked
// [bh][s/32][dh][s%32] so the ekfv kernel can async-stage contiguous B tiles.
__global__ __launch_bounds__(256) void proj_kernel(
    const bf16* __restrict__ hsb,
    const bf16* __restrict__ wqB, const bf16* __restrict__ wkB,
    const bf16* __restrict__ wvB,
    const float* __restrict__ bq, const float* __restrict__ bk,
    const float* __restrict__ bv,
    const unsigned char* __restrict__ mask,
    bf16* __restrict__ qo, bf16* __restrict__ ko, bf16* __restrict__ vo) {
  __shared__ bf16 Bs[2][64 * 32];   // double-buffered B tile [n][k], 4KB each

  const int z = blockIdx.z;
  const bf16*  WB   = (z == 0) ? wqB : (z == 1) ? wkB : wvB;
  const float* bias = (z == 0) ? bq  : (z == 1) ? bk  : bv;
  bf16*        out  = (z == 0) ? qo  : (z == 1) ? ko  : vo;

  const int tid   = threadIdx.x;
  const int wave  = tid >> 5;
  const int lane  = tid & 31;
  const int laneM = lane & 15;
  const int hi    = lane >> 4;
  const int akb   = hi * 8;     // A-fragment K base (ISA 16-bit A layout)
  const int bks   = hi * 8;     // B-fragment K base (same order as A)

  const int rowBase = blockIdx.x * 128 + wave * 16;
  const int n0      = blockIdx.y * 64;
  const int aRow    = rowBase + laneM;

  const unsigned lds0 = (unsigned)(uintptr_t)(&Bs[0][0]) + tid * 16;
  const unsigned lds1 = lds0 + 4096;
  // tile kt lives at WB + (kt*1024 + n0)*32 elements, 4KB contiguous
  const bf16* gtile0 = WB + ((size_t)n0 << 5) + tid * 8;

  constexpr int NT = CD / 32;          // 32 k-tiles (even)
  constexpr size_t TSTEP = (size_t)CD * 32;  // elements between k-tiles
  v8f acc[4] = {};

  auto compute = [&](const bf16* bsBase, int kt) {
    const bf16* ap = hsb + (size_t)aRow * CD + kt * 32 + akb;
    __builtin_prefetch(ap + 64, 0, 1);       // global_prefetch_b8
    v16bf af = load_frag(ap, ap + 16);
    for (int s = 0; s < 4; ++s) {
      const bf16* bp = bsBase + (s * 16 + laneM) * 32 + bks;
      v16bf bfv = load_frag(bp, bp + 16);
      acc[s] = wmma_bf16(af, bfv, acc[s]);
    }
  };

  async_copy_b128(lds0, gtile0);             // prefill buffer 0 (kt = 0)
  for (int kt = 0; kt < NT; kt += 2) {
    // even tile: buf0 compute, buf1 in flight
    async_copy_b128(lds1, gtile0 + (size_t)(kt + 1) * TSTEP);
    ASYNC_WAIT(1);                           // tile kt landed
    __syncthreads();
    compute(&Bs[0][0], kt);
    __syncthreads();                         // reads done before next DMA
    // odd tile: buf1 compute, buf0 in flight (except at tail)
    if (kt + 2 < NT) {
      async_copy_b128(lds0, gtile0 + (size_t)(kt + 2) * TSTEP);
      ASYNC_WAIT(1);
    } else {
      ASYNC_WAIT(0);
    }
    __syncthreads();
    compute(&Bs[1][0], kt + 1);
    __syncthreads();
  }

  // epilogue: bias + mask + scatter
  for (int s = 0; s < 4; ++s) {
    for (int j = 0; j < 8; ++j) {
      int m = rowBase + hi * 8 + j;          // token index
      int c = n0 + s * 16 + laneM;           // model-dim column
      float r = acc[s][j] + bias[c];
      if (!mask[m]) r = 0.0f;
      int b  = m >> 12;                      // /S
      int ss = m & (CS - 1);
      int h  = c >> 6;                       // /DH
      int dh = c & (CDH - 1);
      size_t base = (size_t)(b * CH + h) * CS * CDH;
      if (z == 0)
        out[base + (size_t)ss * CDH + dh] = (bf16)r;                 // row-major
      else
        out[base + ((size_t)(ss >> 5) << 11) + (dh << 5) + (ss & 31)] = (bf16)r;
    }
  }
}

// ---------------- stage 2: E/F sequence projections ----------------
// z==0: ek [bh][m][d] = sum_s We[m,s]*k[bh][s,d] + be[m]   (row-major KPxDH)
// z==1: fvT[bh][d][m] = sum_s Wf[m,s]*v[bh][s,d] + bf[m]   (transposed DHxKP)
// k/v arrive k-blocked: [bh][s/32][dh][s%32] -> contiguous 4KB async B tiles.
__global__ __launch_bounds__(256) void ekfv_kernel(
    const bf16* __restrict__ web, const bf16* __restrict__ wfb,
    const bf16* __restrict__ kbuf, const bf16* __restrict__ vbuf,
    const float* __restrict__ be, const float* __restrict__ bfb,
    bf16* __restrict__ ek, bf16* __restrict__ fvT) {
  __shared__ bf16 Bs[2][64 * 32];

  const int z  = blockIdx.z;
  const int bh = blockIdx.y;
  const bf16*  A    = (z == 0) ? web : wfb;
  const bf16*  kv   = ((z == 0) ? kbuf : vbuf) + (size_t)bh * CS * CDH;
  const float* bias = (z == 0) ? be : bfb;

  const int tid   = threadIdx.x;
  const int wave  = tid >> 5;
  const int lane  = tid & 31;
  const int laneM = lane & 15;
  const int hi    = lane >> 4;
  const int akb   = hi * 8;
  const int bks   = hi * 8;
  const int rowBase = blockIdx.x * 128 + wave * 16;

  const unsigned lds0 = (unsigned)(uintptr_t)(&Bs[0][0]) + tid * 16;
  const unsigned lds1 = lds0 + 4096;
  const bf16* gtile0 = kv + tid * 8;   // tile kt at kv + kt*2048 elements

  constexpr int NT = CS / 32;          // 128 s-tiles (even)
  v8f acc[4] = {};

  auto compute = [&](const bf16* bsBase, int kt) {
    const bf16* ap = A + (size_t)(rowBase + laneM) * CS + kt * 32 + akb;
    __builtin_prefetch(ap + 64, 0, 1);
    v16bf af = load_frag(ap, ap + 16);
    for (int s = 0; s < 4; ++s) {
      const bf16* bp = bsBase + (s * 16 + laneM) * 32 + bks;
      v16bf bfv = load_frag(bp, bp + 16);
      acc[s] = wmma_bf16(af, bfv, acc[s]);
    }
  };

  async_copy_b128(lds0, gtile0);
  for (int kt = 0; kt < NT; kt += 2) {
    async_copy_b128(lds1, gtile0 + (size_t)(kt + 1) * 2048);
    ASYNC_WAIT(1);
    __syncthreads();
    compute(&Bs[0][0], kt);
    __syncthreads();
    if (kt + 2 < NT) {
      async_copy_b128(lds0, gtile0 + (size_t)(kt + 2) * 2048);
      ASYNC_WAIT(1);
    } else {
      ASYNC_WAIT(0);
    }
    __syncthreads();
    compute(&Bs[1][0], kt + 1);
    __syncthreads();
  }

  for (int s = 0; s < 4; ++s) {
    for (int j = 0; j < 8; ++j) {
      int m = rowBase + hi * 8 + j;   // linformer-K row
      int c = s * 16 + laneM;         // head-dim column
      float r = acc[s][j] + bias[m];
      if (z == 0)
        ek[(size_t)bh * CKP * CDH + (size_t)m * CDH + c] = (bf16)r;
      else
        fvT[(size_t)bh * CDH * CKP + (size_t)c * CKP + m] = (bf16)r;
    }
  }
}

// ---------------- stage 3: fused scores -> softmax -> ctx ----------------
__global__ __launch_bounds__(256) void attn_kernel(
    const bf16* __restrict__ q, const bf16* __restrict__ ek,
    const bf16* __restrict__ fvT, float* __restrict__ out) {
  __shared__ bf16 Pch[8][16 * 32];   // per-wave prob chunk (16 rows x 32 k)

  const int wave  = threadIdx.x >> 5;
  const int lane  = threadIdx.x & 31;
  const int laneM = lane & 15;
  const int hi    = lane >> 4;
  const int akb   = hi * 8;
  const int bks   = hi * 8;

  const int bh    = blockIdx.y;
  const int sBase = blockIdx.x * 128 + wave * 16;
  const bf16* qb  = q   + (size_t)bh * CS  * CDH;
  const bf16* ekb = ek  + (size_t)bh * CKP * CDH;
  const bf16* fvb = fvT + (size_t)bh * CDH * CKP;

  // ---- scores: 16 column subtiles of 16, inner dim DH=64 (2 k-steps)
  v8f sc[16] = {};
  for (int kt = 0; kt < 2; ++kt) {
    const bf16* ap = qb + (size_t)(sBase + laneM) * CDH + kt * 32 + akb;
    v16bf af = load_frag(ap, ap + 16);
    for (int sub = 0; sub < 16; ++sub) {
      const bf16* bp = ekb + (size_t)(sub * 16 + laneM) * CDH + kt * 32 + bks;
      v16bf bfv = load_frag(bp, bp + 16);
      sc[sub] = wmma_bf16(af, bfv, sc[sub]);
    }
  }

  // ---- softmax over 256 cols; row r = sBase + hi*8 + j lives in one
  // 16-lane half-wave (same hi), element j, spread over 16 subtiles.
  const float scale = 0.125f;   // 1/sqrt(64)
  for (int j = 0; j < 8; ++j) {
    float mx = -3.4e38f;
    for (int sub = 0; sub < 16; ++sub) {
      float v = sc[sub][j] * scale;
      sc[sub][j] = v;
      mx = fmaxf(mx, v);
    }
    mx = fmaxf(mx, __shfl_xor(mx, 1, 32));
    mx = fmaxf(mx, __shfl_xor(mx, 2, 32));
    mx = fmaxf(mx, __shfl_xor(mx, 4, 32));
    mx = fmaxf(mx, __shfl_xor(mx, 8, 32));
    float sum = 0.0f;
    for (int sub = 0; sub < 16; ++sub) {
      float e = __expf(sc[sub][j] - mx);
      sc[sub][j] = e;
      sum += e;
    }
    sum += __shfl_xor(sum, 1, 32);
    sum += __shfl_xor(sum, 2, 32);
    sum += __shfl_xor(sum, 4, 32);
    sum += __shfl_xor(sum, 8, 32);
    float inv = 1.0f / sum;
    for (int sub = 0; sub < 16; ++sub) sc[sub][j] *= inv;
  }

  // ---- ctx = P(16x256) x FV(256x64), 8 k-steps of 32
  v8f cx[4] = {};
  for (int kt2 = 0; kt2 < 8; ++kt2) {
    __syncthreads();   // WAR guard on Pch
    for (int u = 0; u < 2; ++u) {
      int sub = kt2 * 2 + u;
      for (int j = 0; j < 8; ++j)
        Pch[wave][(hi * 8 + j) * 32 + u * 16 + laneM] = (bf16)sc[sub][j];
    }
    __syncthreads();   // RAW guard
    const bf16* pp = &Pch[wave][laneM * 32 + akb];
    v16bf af = load_frag(pp, pp + 16);
    for (int s2 = 0; s2 < 4; ++s2) {
      const bf16* bp = fvb + (size_t)(s2 * 16 + laneM) * CKP + kt2 * 32 + bks;
      v16bf bfv = load_frag(bp, bp + 16);
      cx[s2] = wmma_bf16(af, bfv, cx[s2]);
    }
  }

  // ---- output [B,S,D] fp32
  const int b = bh >> 4;
  const int h = bh & 15;
  for (int s2 = 0; s2 < 4; ++s2) {
    for (int j = 0; j < 8; ++j) {
      int srow = sBase + hi * 8 + j;
      int d = s2 * 16 + laneM;
      out[(size_t)(b * CS + srow) * CD + h * CDH + d] = cx[s2][j];
    }
  }
}

// ---------------- host launch ----------------
extern "C" void kernel_launch(void* const* d_in, const int* in_sizes, int n_in,
                              void* d_out, int out_size, void* d_ws, size_t ws_size,
                              hipStream_t stream) {
  const float*         hs   = (const float*)d_in[0];
  const unsigned char* mask = (const unsigned char*)d_in[1];  // bool [B,1,S,1]
  const float* Wq = (const float*)d_in[2];
  const float* bq = (const float*)d_in[3];
  const float* Wk = (const float*)d_in[4];
  const float* bk = (const float*)d_in[5];
  const float* Wv = (const float*)d_in[6];
  const float* bv = (const float*)d_in[7];
  const float* We = (const float*)d_in[8];
  const float* be = (const float*)d_in[9];
  const float* Wf = (const float*)d_in[10];
  const float* bf = (const float*)d_in[11];
  float* out = (float*)d_out;

  char* ws = (char*)d_ws;
  size_t off = 0;
  auto alloc = [&](size_t bytes) {
    char* p = ws + off;
    off += (bytes + 255) & ~(size_t)255;
    return p;
  };
  bf16* hsb  = (bf16*)alloc((size_t)NTOK * CD * sizeof(bf16));   // 32 MB
  bf16* wqB  = (bf16*)alloc((size_t)CD * CD * sizeof(bf16));     // 2 MB each
  bf16* wkB  = (bf16*)alloc((size_t)CD * CD * sizeof(bf16));
  bf16* wvB  = (bf16*)alloc((size_t)CD * CD * sizeof(bf16));
  bf16* web  = (bf16*)alloc((size_t)CKP * CS * sizeof(bf16));    // 2 MB each
  bf16* wfb  = (bf16*)alloc((size_t)CKP * CS * sizeof(bf16));
  bf16* qbuf = (bf16*)alloc((size_t)CB * CH * CS * CDH * sizeof(bf16)); // 32 MB each
  bf16* kbuf = (bf16*)alloc((size_t)CB * CH * CS * CDH * sizeof(bf16));
  bf16* vbuf = (bf16*)alloc((size_t)CB * CH * CS * CDH * sizeof(bf16));
  bf16* ekb  = (bf16*)alloc((size_t)CB * CH * CKP * CDH * sizeof(bf16)); // 2 MB each
  bf16* fvTb = (bf16*)alloc((size_t)CB * CH * CDH * CKP * sizeof(bf16));
  (void)ws_size;

  // stage 0: casts / blocked transposes
  cast_bf16_kernel<<<4096, 256, 0, stream>>>(hs, hsb, NTOK * CD);
  transpose_block_kernel<<<(CD * CD) / 256, 256, 0, stream>>>(Wq, wqB);
  transpose_block_kernel<<<(CD * CD) / 256, 256, 0, stream>>>(Wk, wkB);
  transpose_block_kernel<<<(CD * CD) / 256, 256, 0, stream>>>(Wv, wvB);
  cast_bf16_kernel<<<2048, 256, 0, stream>>>(We, web, CKP * CS);
  cast_bf16_kernel<<<2048, 256, 0, stream>>>(Wf, wfb, CKP * CS);

  // stage 1: QKV projections (z selects weight), masked
  proj_kernel<<<dim3(NTOK / 128, CD / 64, 3), 256, 0, stream>>>(
      hsb, wqB, wkB, wvB, bq, bk, bv, mask, qbuf, kbuf, vbuf);

  // stage 2: ek (row-major) and fvT (transposed), per (b,h)
  ekfv_kernel<<<dim3(CKP / 128, CB * CH, 2), 256, 0, stream>>>(
      web, wfb, kbuf, vbuf, be, bf, ekb, fvTb);

  // stage 3: fused scores/softmax/ctx -> fp32 [B,S,D]
  attn_kernel<<<dim3(CS / 128, CB * CH), 256, 0, stream>>>(qbuf, ekb, fvTb, out);
}
